// F_attn_14353780703495
// MI455X (gfx1250) — matrix operand order repaired and checked by reading, not compiled, not run on hardware
//
#include <hip/hip_runtime.h>

// Problem constants (from reference): B=64, S=128, E=256, D=128, N=64*64=4096
#define BB 64
#define SS 128
#define EE 256
#define DD 128
#define NNL 4096
#define NT 64           // n-tile per workgroup in kernel 2

// LDS strides padded to kill bank conflicts
#define ST_STRIDE 133   // f32 elements per column of sT
#define HV_STRIDE 136   // bf16 elements per row of hvT/betaT

typedef __attribute__((ext_vector_type(16))) __bf16 v16bf;
typedef __attribute__((ext_vector_type(8)))  float  v8f;

union ABf { v16bf v; uint4 q[2]; };

__device__ __forceinline__ v8f wmma_bf16(const ABf& a, const ABf& b, v8f c) {
  // D = A(16x32 bf16) * B(32x16 bf16) + C(16x16 f32)
  return __builtin_amdgcn_wmma_f32_16x16x32_bf16(
      /*neg_a=*/false, a.v, /*neg_b=*/false, b.v,
      /*c_mod=*/(short)0, c, /*reuse_a=*/false, /*reuse_b=*/false);
}

// ---------------------------------------------------------------------------
// Kernel 1: ep[b,s,d] = sum_k e[b,s,k] * W[d,k] + bias[d], written as bf16
// both row-major (ep) and transposed (epT). One WG per batch; wave w owns
// M-tile w (16 rows of S), all 8 N-tiles of D.
// ---------------------------------------------------------------------------
__global__ __launch_bounds__(256) void ep_proj_kernel(
    const float* __restrict__ e, const float* __restrict__ Wm,
    const float* __restrict__ bias,
    __bf16* __restrict__ epw, __bf16* __restrict__ epTw)
{
  const int b    = blockIdx.x;
  const int t    = threadIdx.x;
  const int lane = t & 31;
  const int w    = t >> 5;        // wave id 0..7 -> M tile
  const int half = lane >> 4;     // 0: K lo chunks, 1: K hi chunks
  const int m16  = lane & 15;

  v8f acc[8];
#pragma unroll
  for (int j = 0; j < 8; ++j) acc[j] = (v8f)(0.0f);

  const float* erow = e + ((size_t)b * SS + w * 16 + m16) * EE;

  for (int k0 = 0; k0 < EE; k0 += 32) {
    // A tile: row = w*16+m16, K = k0+half*16 .. +15
    float fa[16];
    *(float4*)&fa[0]  = *(const float4*)(erow + k0 + half * 16);
    *(float4*)&fa[4]  = *(const float4*)(erow + k0 + half * 16 + 4);
    *(float4*)&fa[8]  = *(const float4*)(erow + k0 + half * 16 + 8);
    *(float4*)&fa[12] = *(const float4*)(erow + k0 + half * 16 + 12);
    ABf a;
#pragma unroll
    for (int i = 0; i < 16; ++i) a.v[i] = (__bf16)fa[i];

#pragma unroll
    for (int j = 0; j < 8; ++j) {
      // B tile: column d = j*16+m16 is W row d; 16 consecutive k
      const float* wrow = Wm + (size_t)(j * 16 + m16) * EE + k0 + half * 16;
      float fb[16];
      *(float4*)&fb[0]  = ((const float4*)wrow)[0];
      *(float4*)&fb[4]  = ((const float4*)wrow)[1];
      *(float4*)&fb[8]  = ((const float4*)wrow)[2];
      *(float4*)&fb[12] = ((const float4*)wrow)[3];
      ABf bm;
#pragma unroll
      for (int i = 0; i < 16; ++i) bm.v[i] = (__bf16)fb[i];
      acc[j] = wmma_bf16(a, bm, acc[j]);
    }
  }

  __bf16* epb  = epw  + (size_t)b * SS * DD;
  __bf16* eptb = epTw + (size_t)b * DD * SS;
#pragma unroll
  for (int j = 0; j < 8; ++j) {
    const int d  = j * 16 + m16;
    const float bb = bias[d];
#pragma unroll
    for (int r = 0; r < 8; ++r) {
      const int s = w * 16 + r + half * 8;
      const __bf16 vv = (__bf16)(acc[j][r] + bb);
      epb[(size_t)s * DD + d] = vv;
      eptb[(size_t)d * SS + s] = vv;
    }
  }
}

// ---------------------------------------------------------------------------
// Kernel 2: fused  s = ep @ hv_tile  ->  softmax over S  ->  c = epT @ beta.
// One WG per (batch, 64-wide n chunk). Whole softmax axis (S=128) lives in
// LDS; beta never touches HBM. 1/sum folded into the output store.
// ---------------------------------------------------------------------------
__global__ __launch_bounds__(256) void attn_fused_kernel(
    const float* __restrict__ h, const __bf16* __restrict__ epw,
    const __bf16* __restrict__ epTw, float* __restrict__ out)
{
  __shared__ float  sT[NT * ST_STRIDE];      // s tile, transposed (n-major)
  __shared__ __bf16 hvT[NT * HV_STRIDE];     // hv tile bf16 transposed; reused as betaT
  __shared__ float  red[4 * NT];
  __shared__ float  colmax[NT];
  __shared__ float  colrecip[NT];

  const int nc   = blockIdx.x;
  const int b    = blockIdx.y;
  const int n0   = nc * NT;
  const int t    = threadIdx.x;
  const int lane = t & 31;
  const int w    = t >> 5;
  const int half = lane >> 4;
  const int m16  = lane & 15;

  // ---- stage hv tile (D x NT f32) -> LDS transposed bf16 (coalesced reads)
  const float* hsrc = h + (size_t)b * DD * NNL + n0;
  for (int idx = t; idx < DD * NT; idx += 256) {
    const int d = idx >> 6;
    const int c = idx & 63;
    hvT[c * HV_STRIDE + d] = (__bf16)hsrc[(size_t)d * NNL + c];
  }
  __syncthreads();

  // ---- s = ep @ hv : M=S(8 waves x 16), N=NT(4 tiles), K=D=128
  const __bf16* epb = epw + (size_t)b * SS * DD + (size_t)(w * 16 + m16) * DD;
  v8f acc[4];
#pragma unroll
  for (int j = 0; j < 4; ++j) acc[j] = (v8f)(0.0f);
#pragma unroll
  for (int k0 = 0; k0 < DD; k0 += 32) {
    ABf a;
    a.q[0] = *(const uint4*)(epb + k0 + half * 16);
    a.q[1] = *(const uint4*)(epb + k0 + half * 16 + 8);
#pragma unroll
    for (int j = 0; j < 4; ++j) {
      const __bf16* p = &hvT[(j * 16 + m16) * HV_STRIDE + k0 + half * 16];
      ABf bm;
      bm.q[0] = ((const uint4*)p)[0];
      bm.q[1] = ((const uint4*)p)[1];
      acc[j] = wmma_bf16(a, bm, acc[j]);
    }
  }
#pragma unroll
  for (int j = 0; j < 4; ++j)
#pragma unroll
    for (int r = 0; r < 8; ++r)
      sT[(j * 16 + m16) * ST_STRIDE + w * 16 + r + half * 8] = acc[j][r];
  __syncthreads();

  // ---- softmax over S per column n (column = contiguous LDS row of sT)
  {
    const int col = t & 63;
    const int q   = t >> 6;                  // quarter of the S axis
    const float* src = &sT[col * ST_STRIDE + q * 32];
    float m = -3.0e38f;
#pragma unroll
    for (int i = 0; i < 32; ++i) m = fmaxf(m, src[i]);
    red[q * NT + col] = m;
    __syncthreads();
    if (t < NT)
      colmax[t] = fmaxf(fmaxf(red[t], red[NT + t]),
                        fmaxf(red[2 * NT + t], red[3 * NT + t]));
    __syncthreads();
    const float mx = colmax[col];
    __bf16* bdst = &hvT[col * HV_STRIDE + q * 32];   // betaT overwrites hvT
    float ssum = 0.0f;
#pragma unroll
    for (int i = 0; i < 32; ++i) {
      const float vv = __expf(src[i] - mx);
      ssum += vv;
      bdst[i] = (__bf16)vv;
    }
    red[q * NT + col] = ssum;
    __syncthreads();
    if (t < NT)
      colrecip[t] = 1.0f / (red[t] + red[NT + t] + red[2 * NT + t] + red[3 * NT + t]);
    __syncthreads();
  }

  // ---- c = epT @ beta : M=D(8 waves x 16), N=NT(4 tiles), K=S=128
  const __bf16* eptb = epTw + (size_t)b * DD * SS + (size_t)(w * 16 + m16) * SS;
#pragma unroll
  for (int j = 0; j < 4; ++j) acc[j] = (v8f)(0.0f);
#pragma unroll
  for (int k0 = 0; k0 < SS; k0 += 32) {
    ABf a;
    a.q[0] = *(const uint4*)(eptb + k0 + half * 16);
    a.q[1] = *(const uint4*)(eptb + k0 + half * 16 + 8);
#pragma unroll
    for (int j = 0; j < 4; ++j) {
      const __bf16* p = &hvT[(j * 16 + m16) * HV_STRIDE + k0 + half * 16];
      ABf bm;
      bm.q[0] = ((const uint4*)p)[0];
      bm.q[1] = ((const uint4*)p)[1];
      acc[j] = wmma_bf16(a, bm, acc[j]);
    }
  }

  float* outp = out + (size_t)b * DD * NNL + n0;
#pragma unroll
  for (int j = 0; j < 4; ++j) {
    const int   nn  = j * 16 + m16;
    const float rcp = colrecip[nn];
#pragma unroll
    for (int r = 0; r < 8; ++r) {
      const int dd = w * 16 + r + half * 8;
      outp[(size_t)dd * NNL + nn] = acc[j][r] * rcp;
    }
  }
}

// ---------------------------------------------------------------------------
// inputs: e (B,S,E) f32 | h (B,D,64,64) f32 | W (D,E) f32 | b (D,) f32
// out: (B,D,64,64) f32.  Workspace: ep bf16 (2 MB) + epT bf16 (2 MB).
// ---------------------------------------------------------------------------
extern "C" void kernel_launch(void* const* d_in, const int* in_sizes, int n_in,
                              void* d_out, int out_size, void* d_ws, size_t ws_size,
                              hipStream_t stream) {
  const float* e    = (const float*)d_in[0];
  const float* h    = (const float*)d_in[1];
  const float* Wm   = (const float*)d_in[2];
  const float* bias = (const float*)d_in[3];
  float* out = (float*)d_out;

  __bf16* epw  = (__bf16*)d_ws;
  __bf16* epTw = epw + (size_t)BB * SS * DD;   // needs 2 * 2 MB = 4 MB total

  ep_proj_kernel<<<dim3(BB), dim3(256), 0, stream>>>(e, Wm, bias, epw, epTw);
  attn_fused_kernel<<<dim3(NNL / NT, BB), dim3(256), 0, stream>>>(h, epw, epTw, out);
  (void)in_sizes; (void)n_in; (void)out_size; (void)ws_size;
}
